// MoE_51067161149790
// MI455X (gfx1250) — matrix operand order repaired and checked by reading, not compiled, not run on hardware
//
#include <hip/hip_runtime.h>
#include <hip/hip_bf16.h>

// ---------------- problem constants ----------------
#define TOK   2048      // B*S
#define DIM   2048
#define NEXP  32
#define INTER 1024
#define SI    2048      // shared inter
#define NGRP  8
#define TOPKG 4
#define TOPK  6
#define RSCALE 2.5f

typedef __attribute__((ext_vector_type(16))) __bf16 v16bf;
typedef __attribute__((ext_vector_type(8)))  float  v8f;
typedef __attribute__((ext_vector_type(2)))  float  f32x2;
typedef __attribute__((ext_vector_type(2)))  __bf16 bf16x2;

static __device__ __forceinline__ v8f wmma_bf16(v16bf a, v16bf b, v8f c) {
  return __builtin_amdgcn_wmma_f32_16x16x32_bf16(
      false, a, false, b, (short)0, c, false, false);
}

static __device__ __forceinline__ unsigned pack_bf2(float a, float b) {
  f32x2 f = {a, b};
  bf16x2 h = __builtin_convertvector(f, bf16x2);   // v_cvt_pk_bf16_f32 if available
  union { bf16x2 v; unsigned u; } t;
  t.v = h;
  return t.u;
}

// silu via fast v_rcp_f32 instead of IEEE divide chains
static __device__ __forceinline__ float silu(float g) {
  return g * __builtin_amdgcn_rcpf(1.0f + __expf(-g));
}

// ---- CDNA5 async global->LDS copy (16B per lane), tracked by ASYNCcnt -------
static __device__ __forceinline__ void async_copy_b128(unsigned lds_byte_addr,
                                                       const void* gptr) {
  asm volatile("global_load_async_to_lds_b128 %0, %1, off"
               :: "v"(lds_byte_addr), "v"((unsigned long long)(uintptr_t)gptr)
               : "memory");
}
static __device__ __forceinline__ void wait_async0() {
  asm volatile("s_wait_asynccnt 0x0" ::: "memory");
}

// ---- fragment-ordered LDS staging -------------------------------------------
// B tile: 64 N rows x 32 K fp32 -> 4 subtiles of [lane][16] bf16; a B-fragment
// is one contiguous 32B chunk per lane. task in [0,256): n=task>>2, q=task&3.
static __device__ __forceinline__ void stage_w(const float* __restrict__ W,
                                               size_t gbase, int ld, int kk,
                                               __bf16* __restrict__ Bs, int task) {
  int n = task >> 2, q = task & 3;
  const float* s = W + gbase + (size_t)n * ld + (kk + q * 8);
  float4 f0 = *(const float4*)s;
  float4 f1 = *(const float4*)(s + 4);
  __builtin_prefetch((const void*)(s + 32), 0, 0);   // next K slab -> global_prefetch
  uint4 p;
  p.x = pack_bf2(f0.x, f0.y); p.y = pack_bf2(f0.z, f0.w);
  p.z = pack_bf2(f1.x, f1.y); p.w = pack_bf2(f1.z, f1.w);
  int lane = (n & 15) + ((q & 1) << 4);
  *(uint4*)&Bs[(n >> 4) * 512 + lane * 16 + ((q >> 1) << 3)] = p;
}

// A tile staging via async DMA: row r, 8-wide chunk c8; 16B copy, frag-ordered.
// lds_base = byte offset of the A region within the (dynamic) LDS segment.
static __device__ __forceinline__ void stage_a_async(const __bf16* __restrict__ X,
                                                     size_t row, int ld, int kk,
                                                     unsigned lds_base,
                                                     int r, int c8) {
  const void* g = &X[row * (size_t)ld + kk + c8 * 8];
  int lane = (r & 15) + ((c8 & 1) << 4);
  unsigned off = lds_base +
      2u * (unsigned)((r >> 4) * 512 + lane * 16 + ((c8 >> 1) << 3));
  async_copy_b128(off, g);
}

static __device__ __forceinline__ v16bf ld_frag(const __bf16* __restrict__ p,
                                                int tile, int lane) {
  return *(const v16bf*)&p[tile * 512 + lane * 16];   // 32B aligned
}

// ---------------- kernel 0: zero expert counters ----------------
__global__ void moe_zero_counts(int* counts) {
  if (threadIdx.x < NEXP) counts[threadIdx.x] = 0;
}

// ---------------- kernel 1: x fp32 -> bf16 scratch (8 elems/thread) ----------
__global__ void moe_cvt_x(const float* __restrict__ x, __bf16* __restrict__ xb) {
  int i = blockIdx.x * blockDim.x + threadIdx.x;      // TOK*DIM/8 threads
  const float* s = x + (size_t)i * 8;
  float4 f0 = *(const float4*)s;
  float4 f1 = *(const float4*)(s + 4);
  uint4 p;
  p.x = pack_bf2(f0.x, f0.y); p.y = pack_bf2(f0.z, f0.w);
  p.z = pack_bf2(f1.x, f1.y); p.w = pack_bf2(f1.z, f1.w);
  ((uint4*)xb)[i] = p;
}

// ---------------- kernel 2: gate + group-limited top-k routing ---------------
__global__ void moe_gate(const float* __restrict__ x, const float* __restrict__ gw,
                         int* __restrict__ counts, int* __restrict__ tok_idx,
                         float* __restrict__ tok_w) {
  int t = blockIdx.x;
  int lane = threadIdx.x;           // 0..31, lane == expert
  const float4* xr = (const float4*)(x + (size_t)t * DIM);
  const float4* wr = (const float4*)(gw + (size_t)lane * DIM);
  float s = 0.f;
  for (int d = 0; d < DIM / 4; ++d) {
    float4 a = xr[d], b = wr[d];
    s += a.x * b.x + a.y * b.y + a.z * b.z + a.w * b.w;
  }
  float sc = __builtin_amdgcn_rcpf(1.0f + __expf(-s));   // sigmoid
  __shared__ float ssc[32];
  ssc[lane] = sc;
  __syncthreads();
  if (lane == 0) {
    float gm[NGRP];
#pragma unroll 1
    for (int g = 0; g < NGRP; ++g) {
      float m = ssc[g * 4];
      for (int j = 1; j < 4; ++j) m = fmaxf(m, ssc[g * 4 + j]);
      gm[g] = m;
    }
    int allowed[NGRP];
    for (int g = 0; g < NGRP; ++g) allowed[g] = 0;
#pragma unroll 1
    for (int it = 0; it < TOPKG; ++it) {
      int best = 0; float bv = -1e30f;
#pragma unroll 1
      for (int g = 0; g < NGRP; ++g)
        if (gm[g] > bv) { bv = gm[g]; best = g; }
      allowed[best] = 1; gm[best] = -1e30f;
    }
    float s2[NEXP];
#pragma unroll 1
    for (int e = 0; e < NEXP; ++e) s2[e] = allowed[e >> 2] ? ssc[e] : -1e30f;
    int sel[TOPK]; float sv[TOPK]; float wsum = 0.f;
#pragma unroll 1
    for (int it = 0; it < TOPK; ++it) {
      int best = 0; float bv = -1e30f;
#pragma unroll 1
      for (int e = 0; e < NEXP; ++e)
        if (s2[e] > bv) { bv = s2[e]; best = e; }
      sel[it] = best; sv[it] = ssc[best]; wsum += ssc[best];
      s2[best] = -1e30f;
    }
    float inv = RSCALE * __builtin_amdgcn_rcpf(wsum);
#pragma unroll 1
    for (int it = 0; it < TOPK; ++it) {
      int e = sel[it];
      int pos = atomicAdd(&counts[e], 1);
      tok_idx[e * TOK + pos] = t;
      tok_w[e * TOK + pos] = sv[it] * inv;
    }
  }
}

// ---------------- kernel 3: shared up: h = silu(x@sw1^T)*(x@sw3^T) -----------
// grid (TOK/64, SI/64), 256 threads, dynamic LDS (As@0, B1s@4096, B3s@8192)
__global__ __launch_bounds__(256)
void moe_shared_up(const __bf16* __restrict__ xb, const float* __restrict__ sw1,
                   const float* __restrict__ sw3, __bf16* __restrict__ h) {
  extern __shared__ char smem[];
  __bf16* As  = (__bf16*)smem;       // 4*512 halfs, byte 0
  __bf16* B1s = As + 2048;           // byte 4096
  __bf16* B3s = B1s + 2048;          // byte 8192
  int tid = threadIdx.x, lane = tid & 31, wv = tid >> 5;
  int m0 = blockIdx.x * 64, n0 = blockIdx.y * 64;
  int m_t = wv >> 1;           // 0..3
  int npair = wv & 1;          // subtiles s = npair*2 + {0,1}
  v8f accg[2] = {}, accu[2] = {};
  for (int kk = 0; kk < DIM; kk += 32) {
    stage_a_async(xb, (size_t)(m0 + (tid >> 2)), DIM, kk, 0u, tid >> 2, tid & 3);
    stage_w(sw1, (size_t)n0 * DIM, DIM, kk, B1s, tid);
    stage_w(sw3, (size_t)n0 * DIM, DIM, kk, B3s, tid);
    wait_async0();
    __syncthreads();
    v16bf a = ld_frag(As, m_t, lane);
#pragma unroll
    for (int p = 0; p < 2; ++p) {
      int s = npair * 2 + p;
      accg[p] = wmma_bf16(a, ld_frag(B1s, s, lane), accg[p]);
      accu[p] = wmma_bf16(a, ld_frag(B3s, s, lane), accu[p]);
    }
    __syncthreads();
  }
#pragma unroll
  for (int p = 0; p < 2; ++p)
#pragma unroll
    for (int r = 0; r < 8; ++r) {
      float act = silu(accg[p][r]) * accu[p][r];
      int m = m0 + m_t * 16 + ((lane >> 4) << 3) + r;
      int n = n0 + (npair * 2 + p) * 16 + (lane & 15);
      h[(size_t)m * SI + n] = (__bf16)act;
    }
}

// ---------------- kernel 4: shared down: out = h @ sw2^T ---------------------
// grid (TOK/64, DIM/64), 256 threads, dynamic LDS (As@0, Bs@4096)
__global__ __launch_bounds__(256)
void moe_shared_down(const __bf16* __restrict__ h, const float* __restrict__ sw2,
                     float* __restrict__ out) {
  extern __shared__ char smem[];
  __bf16* As = (__bf16*)smem;        // byte 0
  __bf16* Bs = As + 2048;            // byte 4096
  int tid = threadIdx.x, lane = tid & 31, wv = tid >> 5;
  int m0 = blockIdx.x * 64, n0 = blockIdx.y * 64;
  int m_t = wv >> 1, npair = wv & 1;
  v8f acc[2] = {};
  for (int kk = 0; kk < SI; kk += 32) {
    stage_a_async(h, (size_t)(m0 + (tid >> 2)), SI, kk, 0u, tid >> 2, tid & 3);
    stage_w(sw2, (size_t)n0 * SI, SI, kk, Bs, tid);
    wait_async0();
    __syncthreads();
    v16bf a = ld_frag(As, m_t, lane);
#pragma unroll
    for (int p = 0; p < 2; ++p)
      acc[p] = wmma_bf16(a, ld_frag(Bs, npair * 2 + p, lane), acc[p]);
    __syncthreads();
  }
#pragma unroll
  for (int p = 0; p < 2; ++p)
#pragma unroll
    for (int r = 0; r < 8; ++r) {
      int m = m0 + m_t * 16 + ((lane >> 4) << 3) + r;
      int n = n0 + (npair * 2 + p) * 16 + (lane & 15);
      out[(size_t)m * DIM + n] = acc[p][r];
    }
}

// ---------------- kernel 5: routed experts, fused ----------------------------
// grid = NEXP*64 (tiles of 32 tokens), 256 threads, dynamic LDS (~74 KB)
// layout (bytes): ActF@0 (65536), As@65536 (2048), B1s@67584 (4096),
//                 B3s@71680 (4096), tks@75776, twt@75904
#define K5_AS_BYTE 65536u
__global__ __launch_bounds__(256)
void moe_routed(const __bf16* __restrict__ xb, const float* __restrict__ w11,
                const float* __restrict__ w33, const float* __restrict__ w22,
                const int* __restrict__ counts, const int* __restrict__ tok_idx,
                const float* __restrict__ tok_w, float* __restrict__ out) {
  extern __shared__ char smem[];
  __bf16* ActF = (__bf16*)smem;          // 32 slabs x 2 tiles x 512, A-frag order
  __bf16* As   = ActF + 32768;
  __bf16* B1s  = As + 1024;
  __bf16* B3s  = B1s + 2048;
  int*    tks  = (int*)(B3s + 2048);     // 32
  float*  twt  = (float*)(tks + 32);     // 32

  int e  = blockIdx.x >> 6;
  int mt = blockIdx.x & 63;
  int cnt = counts[e];
  if (mt * 32 >= cnt) return;            // block-uniform

  int tid = threadIdx.x, lane = tid & 31, wv = tid >> 5;
  if (tid < 32) {
    int slot = mt * 32 + tid;
    tks[tid] = (slot < cnt) ? tok_idx[e * TOK + slot] : 0;
    twt[tid] = (slot < cnt) ? tok_w[e * TOK + slot] : 0.0f;
  }
  __syncthreads();

  int m_t = wv >> 2;      // 0..1
  int n_q = wv & 3;       // 0..3 (B subtile)
  const size_t wbase  = (size_t)e * INTER * DIM;
  const size_t w2base = (size_t)e * DIM * INTER;

  // ---- phase 1: ActF = silu(X@w11^T) * (X@w33^T), stored in A-frag order ----
  for (int nc = 0; nc < INTER / 64; ++nc) {
    v8f accg = {}, accu = {};
    for (int kk = 0; kk < DIM; kk += 32) {
      if (tid < 128) {
        int r = tid >> 2;
        stage_a_async(xb, (size_t)tks[r], DIM, kk, K5_AS_BYTE, r, tid & 3);
      }
      stage_w(w11, wbase + (size_t)(nc * 64) * DIM, DIM, kk, B1s, tid);
      stage_w(w33, wbase + (size_t)(nc * 64) * DIM, DIM, kk, B3s, tid);
      wait_async0();
      __syncthreads();
      v16bf a = ld_frag(As, m_t, lane);
      accg = wmma_bf16(a, ld_frag(B1s, n_q, lane), accg);
      accu = wmma_bf16(a, ld_frag(B3s, n_q, lane), accu);
      __syncthreads();
    }
#pragma unroll
    for (int r = 0; r < 8; ++r) {
      float act = silu(accg[r]) * accu[r];
      int m = m_t * 16 + ((lane >> 4) << 3) + r;
      int n = nc * 64 + n_q * 16 + (lane & 15);
      // scatter into phase-2 A-fragment order: (row=m, k=n)
      int c  = n & 31;
      int lp = (m & 15) + ((c & 8) ? 16 : 0);
      int jp = (c & 7) + ((c >> 4) << 3);
      ActF[((n >> 5) * 2 + (m >> 4)) * 512 + lp * 16 + jp] = (__bf16)act;
    }
  }
  __syncthreads();

  // ---- phase 2: out[tok,:] += tw * (Act @ w22^T) ----
  __bf16* B2s = B1s;   // reuse
  for (int nc = 0; nc < DIM / 64; ++nc) {
    v8f acc = {};
    for (int kk = 0; kk < INTER; kk += 32) {
      stage_w(w22, w2base + (size_t)(nc * 64) * INTER, INTER, kk, B2s, tid);
      __syncthreads();
      v16bf a = ld_frag(ActF, (kk >> 5) * 2 + m_t, lane);
      acc = wmma_bf16(a, ld_frag(B2s, n_q, lane), acc);
      __syncthreads();
    }
#pragma unroll
    for (int r = 0; r < 8; ++r) {
      int m = m_t * 16 + ((lane >> 4) << 3) + r;
      int n = nc * 64 + n_q * 16 + (lane & 15);
      atomicAdd(&out[(size_t)tks[m] * DIM + n], acc[r] * twt[m]);
    }
  }
}

// ---------------- launch ----------------
extern "C" void kernel_launch(void* const* d_in, const int* in_sizes, int n_in,
                              void* d_out, int out_size, void* d_ws, size_t ws_size,
                              hipStream_t stream) {
  const float* x   = (const float*)d_in[0];
  const float* gw  = (const float*)d_in[1];
  const float* w11 = (const float*)d_in[2];
  const float* w33 = (const float*)d_in[3];
  const float* w22 = (const float*)d_in[4];
  const float* sw1 = (const float*)d_in[5];
  const float* sw2 = (const float*)d_in[6];
  const float* sw3 = (const float*)d_in[7];
  float* out = (float*)d_out;

  char* ws = (char*)d_ws;
  int*    counts  = (int*)ws;                                  // 32 ints (256 B)
  int*    tok_idx = (int*)(ws + 256);
  float*  tok_w   = (float*)(ws + 256 + NEXP * TOK * 4);
  size_t  off_h   = 256 + (size_t)NEXP * TOK * 8;              // 256-aligned
  __bf16* hbuf    = (__bf16*)(ws + off_h);                     // TOK*SI bf16
  size_t  off_xb  = off_h + (size_t)TOK * SI * 2;
  __bf16* xb      = (__bf16*)(ws + off_xb);                    // TOK*DIM bf16

  moe_zero_counts<<<1, 32, 0, stream>>>(counts);

  moe_cvt_x<<<(TOK * DIM / 8) / 256, 256, 0, stream>>>(x, xb);

  moe_gate<<<TOK, 32, 0, stream>>>(x, gw, counts, tok_idx, tok_w);

  moe_shared_up<<<dim3(TOK / 64, SI / 64), 256, 12288, stream>>>(xb, sw1, sw3, hbuf);

  moe_shared_down<<<dim3(TOK / 64, DIM / 64), 256, 8192, stream>>>(hbuf, sw2, out);

  size_t smem = (size_t)(32768 + 1024 + 2048 + 2048) * sizeof(__bf16)
              + 32 * sizeof(int) + 32 * sizeof(float);
  moe_routed<<<dim3(NEXP * 64), 256, smem, stream>>>(
      xb, w11, w33, w22, counts, tok_idx, tok_w, out);
}
